// StyleLossW2_41644002902140
// MI455X (gfx1250) — matrix operand order repaired and checked by reading, not compiled, not run on hardware
//
#include <hip/hip_runtime.h>
#include <math.h>

// ---------------------------------------------------------------------------
// CDNA5 (gfx1250) wave32 WMMA types / helpers
// ---------------------------------------------------------------------------
typedef __attribute__((ext_vector_type(16))) __bf16 v16bf;
typedef __attribute__((ext_vector_type(8)))  __bf16 v8bf;
typedef __attribute__((ext_vector_type(4)))  __bf16 v4bf;
typedef __attribute__((ext_vector_type(8)))  float  v8f;

#define EPS_REG 1e-4f
#define LDS_STRIDE 40   // bf16 elems: 80 B row stride -> 16B aligned, conflict-free

static __device__ __forceinline__ v8f wmma_bf16(v16bf a, v16bf b, v8f c) {
  return __builtin_amdgcn_wmma_f32_16x16x32_bf16(false, a, false, b, (short)0, c,
                                                 false, false);
}

static __device__ __forceinline__ v16bf frag_from(const __bf16* p0, const __bf16* p1) {
  v8bf lo = *(const v8bf*)p0;
  v8bf hi = *(const v8bf*)p1;
  return __builtin_shufflevector(lo, hi, 0, 1, 2, 3, 4, 5, 6, 7,
                                 8, 9, 10, 11, 12, 13, 14, 15);
}

static __device__ __forceinline__ v4bf cvt4(float4 f) {
  return (v4bf){(__bf16)f.x, (__bf16)f.y, (__bf16)f.z, (__bf16)f.w};
}

static __device__ __forceinline__ float block_reduce256(float v, float* sm) {
  const int t = threadIdx.x;
  __syncthreads();
  sm[t] = v;
  __syncthreads();
  for (int o = 128; o > 0; o >>= 1) {
    if (t < o) sm[t] += sm[t + o];
    __syncthreads();
  }
  return sm[0];
}

// ---------------------------------------------------------------------------
// Utility kernels
// ---------------------------------------------------------------------------
__global__ __launch_bounds__(256) void zero_kernel(float* __restrict__ p, long long n) {
  long long i = (long long)blockIdx.x * 256 + threadIdx.x;
  if (i < n) p[i] = 0.f;
}

__global__ __launch_bounds__(256) void row_mean_kernel(const float* __restrict__ X,
                                                       float* __restrict__ out,
                                                       long long rowlen, float inv) {
  const long long row = blockIdx.x;
  const float* p = X + row * rowlen;
  float s = 0.f;
  for (long long i = threadIdx.x; i < rowlen; i += 256) s += p[i];
  __shared__ float sm[256];
  float tot = block_reduce256(s, sm);
  if (threadIdx.x == 0) out[row] = tot * inv;
}

// ---------------------------------------------------------------------------
// Big split-K GEMM: out[b] (+)= X_b * X_b^T (raw), f32 atomics across K-chunks.
// 128x128 tile / workgroup; 8 waves as 4(row)x2(col); 32x64 per wave.
// Both LDS tiles K-contiguous per row/col: [idx][k], stride 40 bf16.
// ---------------------------------------------------------------------------
__global__ __launch_bounds__(256) void gemm_aat_kernel(const float* __restrict__ X,
                                                       float* __restrict__ out,
                                                       long long ldk,
                                                       int kc_per_batch, int KC) {
  __shared__ __align__(16) __bf16 As[2][128][LDS_STRIDE];
  __shared__ __align__(16) __bf16 Bs[2][128][LDS_STRIDE];
  const int tid  = threadIdx.x;
  const int wave = tid >> 5;
  const int lane = tid & 31;
  const int bi = blockIdx.x * 128;
  const int bj = blockIdx.y * 128;
  const int b  = blockIdx.z / kc_per_batch;
  const int kc = blockIdx.z % kc_per_batch;
  const long long k0 = (long long)kc * (long long)KC;
  const float* Xb = X + (long long)b * 256LL * ldk;
  float* Ob = out + (long long)b * 65536LL;

  const int rb = (wave & 3) * 32;   // wave row block within 128
  const int cb = (wave >> 2) * 64;  // wave col block within 128
  const int lr = lane & 15;
  const int khalf = lane >> 4;      // 0|1
  const int abase = khalf * 8;      // A frag: K {0..7,16..23} | {8..15,24..31}
  const int bbase = khalf * 8;      // B frag: K {0..7,16..23} | {8..15,24..31}

  v8f acc[2][4];
#pragma unroll
  for (int rt = 0; rt < 2; ++rt)
#pragma unroll
    for (int ct = 0; ct < 4; ++ct)
      acc[rt][ct] = (v8f){0.f, 0.f, 0.f, 0.f, 0.f, 0.f, 0.f, 0.f};

  auto stage = [&](int buf, long long kk) {
#pragma unroll
    for (int r = 0; r < 4; ++r) {
      const int v = tid + 256 * r;       // 0..1023 float4 slots
      const int row = v >> 3;            // 0..127
      const int ko  = (v & 7) * 4;       // 0..28
      float4 a4 = *(const float4*)&Xb[(long long)(bi + row) * ldk + kk + ko];
      *(v4bf*)&As[buf][row][ko] = cvt4(a4);
      float4 b4 = *(const float4*)&Xb[(long long)(bj + row) * ldk + kk + ko];
      *(v4bf*)&Bs[buf][row][ko] = cvt4(b4);
    }
  };

  const int nsteps = KC / 32;
  stage(0, k0);
  int cur = 0;
  for (int s = 0; s < nsteps; ++s) {
    __syncthreads();
    if (s + 1 < nsteps) stage(cur ^ 1, k0 + (long long)(s + 1) * 32);
    if (s + 2 < nsteps)  // gfx1250 global_prefetch_b8 two slabs ahead
      __builtin_prefetch(&Xb[(long long)(bi + (tid >> 1)) * ldk + k0 + (s + 2) * 32], 0, 1);

    v16bf afrag[2];
#pragma unroll
    for (int rt = 0; rt < 2; ++rt) {
      const __bf16* ap = &As[cur][rb + rt * 16 + lr][0];
      afrag[rt] = frag_from(ap + abase, ap + 16 + abase);
    }
#pragma unroll
    for (int ct = 0; ct < 4; ++ct) {
      const __bf16* bp = &Bs[cur][cb + ct * 16 + lr][0];
      v16bf bfrag = frag_from(bp + bbase, bp + 16 + bbase);
#pragma unroll
      for (int rt = 0; rt < 2; ++rt)
        acc[rt][ct] = wmma_bf16(afrag[rt], bfrag, acc[rt][ct]);
    }
    cur ^= 1;
  }

#pragma unroll
  for (int rt = 0; rt < 2; ++rt)
#pragma unroll
    for (int ct = 0; ct < 4; ++ct)
#pragma unroll
      for (int r = 0; r < 8; ++r) {
        const int grow = bi + rb + rt * 16 + khalf * 8 + r;
        const int gcol = bj + cb + ct * 16 + lr;
        atomicAdd(&Ob[(long long)grow * 256 + gcol], acc[rt][ct][r]);
      }
}

// ---------------------------------------------------------------------------
// 256x256x256 GEMM: C[b] = post(A[b]*B[b]).  post: 0 -> P, 1 -> 1.5*I - 0.5*P
// B tile staged transposed into LDS so fragment reads are K-contiguous.
// ---------------------------------------------------------------------------
__global__ __launch_bounds__(256) void gemm256_kernel(const float* __restrict__ A,
                                                      const float* __restrict__ B,
                                                      float* __restrict__ C,
                                                      long long sA, long long sB,
                                                      long long sC, int post) {
  __shared__ __align__(16) __bf16 As[2][128][LDS_STRIDE];
  __shared__ __align__(16) __bf16 Bs[2][128][LDS_STRIDE];
  const int tid  = threadIdx.x;
  const int wave = tid >> 5;
  const int lane = tid & 31;
  const int bi = blockIdx.x * 128;
  const int bj = blockIdx.y * 128;
  const long long b = blockIdx.z;
  const float* Ab = A + b * sA;
  const float* Bb = B + b * sB;
  float* Cb = C + b * sC;

  const int rb = (wave & 3) * 32;
  const int cb = (wave >> 2) * 64;
  const int lr = lane & 15;
  const int khalf = lane >> 4;
  const int abase = khalf * 8;
  const int bbase = khalf * 8;

  v8f acc[2][4];
#pragma unroll
  for (int rt = 0; rt < 2; ++rt)
#pragma unroll
    for (int ct = 0; ct < 4; ++ct)
      acc[rt][ct] = (v8f){0.f, 0.f, 0.f, 0.f, 0.f, 0.f, 0.f, 0.f};

  auto stage = [&](int buf, int kk) {
#pragma unroll
    for (int r = 0; r < 4; ++r) {
      const int v = tid + 256 * r;       // 0..1023
      // A: row-major, K-contiguous float4
      const int arow = v >> 3;
      const int ako  = (v & 7) * 4;
      float4 a4 = *(const float4*)&Ab[(long long)(bi + arow) * 256 + kk + ako];
      *(v4bf*)&As[buf][arow][ako] = cvt4(a4);
      // B: transpose into LDS. 4 coalesced b32 loads (j fastest) -> 1 b64 store
      const int j  = v & 127;
      const int k4 = (v >> 7) * 4;       // 0..28
      float b0 = Bb[(long long)(kk + k4 + 0) * 256 + bj + j];
      float b1 = Bb[(long long)(kk + k4 + 1) * 256 + bj + j];
      float b2 = Bb[(long long)(kk + k4 + 2) * 256 + bj + j];
      float b3 = Bb[(long long)(kk + k4 + 3) * 256 + bj + j];
      *(v4bf*)&Bs[buf][j][k4] = (v4bf){(__bf16)b0, (__bf16)b1, (__bf16)b2, (__bf16)b3};
    }
  };

  stage(0, 0);
  int cur = 0;
#pragma unroll 1
  for (int s = 0; s < 8; ++s) {
    __syncthreads();
    if (s + 1 < 8) stage(cur ^ 1, (s + 1) * 32);

    v16bf afrag[2];
#pragma unroll
    for (int rt = 0; rt < 2; ++rt) {
      const __bf16* ap = &As[cur][rb + rt * 16 + lr][0];
      afrag[rt] = frag_from(ap + abase, ap + 16 + abase);
    }
#pragma unroll
    for (int ct = 0; ct < 4; ++ct) {
      const __bf16* bp = &Bs[cur][cb + ct * 16 + lr][0];
      v16bf bfrag = frag_from(bp + bbase, bp + 16 + bbase);
#pragma unroll
      for (int rt = 0; rt < 2; ++rt)
        acc[rt][ct] = wmma_bf16(afrag[rt], bfrag, acc[rt][ct]);
    }
    cur ^= 1;
  }

#pragma unroll
  for (int rt = 0; rt < 2; ++rt)
#pragma unroll
    for (int ct = 0; ct < 4; ++ct)
#pragma unroll
      for (int r = 0; r < 8; ++r) {
        const int grow = bi + rb + rt * 16 + khalf * 8 + r;
        const int gcol = bj + cb + ct * 16 + lr;
        float v = acc[rt][ct][r];
        if (post == 1) v = 1.5f * ((grow == gcol) ? 1.f : 0.f) - 0.5f * v;
        Cb[(long long)grow * 256 + gcol] = v;
      }
}

// ---------------------------------------------------------------------------
// cov = srm*inv_n - mean(x)mean + eps*I  (in place), plus Frobenius norm.
// ---------------------------------------------------------------------------
__global__ __launch_bounds__(256) void finalize_cov_kernel(float* __restrict__ srm,
                                                           const float* __restrict__ mean,
                                                           float inv_n, float eps,
                                                           float* __restrict__ norm_out) {
  const int b = blockIdx.x;
  float* cov = srm + (long long)b * 65536LL;
  const float* mb = mean + b * 256;
  __shared__ float sm[256];
  sm[threadIdx.x] = mb[threadIdx.x];
  __syncthreads();
  const int row = threadIdx.x;
  const float mr = sm[row];
  float* crow = cov + (long long)row * 256;
  float ss = 0.f;
  for (int c = 0; c < 256; ++c) {
    float v = crow[c] * inv_n - mr * sm[c] + ((c == row) ? eps : 0.f);
    crow[c] = v;
    ss += v * v;
  }
  float tot = block_reduce256(ss, sm);
  if (threadIdx.x == 0) norm_out[b] = sqrtf(tot);
}

__global__ __launch_bounds__(256) void frob_norm_kernel(const float* __restrict__ a,
                                                        float* __restrict__ norm_out) {
  const int b = blockIdx.x;
  const float* p = a + (long long)b * 65536LL;
  float ss = 0.f;
  for (int i = threadIdx.x; i < 65536; i += 256) { float v = p[i]; ss += v * v; }
  __shared__ float sm[256];
  float tot = block_reduce256(ss, sm);
  if (threadIdx.x == 0) norm_out[b] = sqrtf(tot);
}

__global__ __launch_bounds__(256) void init_yz_kernel(const float* __restrict__ a,
                                                      const float* __restrict__ norms,
                                                      float* __restrict__ y,
                                                      float* __restrict__ z) {
  const int b = blockIdx.y;
  const long long i = (long long)blockIdx.x * 256 + threadIdx.x;
  const long long off = (long long)b * 65536LL + i;
  const float nv = norms[b];
  y[off] = a[off] / nv;
  const int r = (int)(i >> 8), c = (int)(i & 255);
  z[off] = (r == c) ? 1.f : 0.f;
}

__global__ __launch_bounds__(256) void scale_copy_kernel(const float* __restrict__ src,
                                                         const float* __restrict__ norms,
                                                         float* __restrict__ dst) {
  const long long i = (long long)blockIdx.x * 256 + threadIdx.x;
  dst[i] = src[i] * sqrtf(norms[0]);
}

__global__ __launch_bounds__(256) void final_kernel(const float* __restrict__ mean_x,
                                                    const float* __restrict__ mean_t,
                                                    const float* __restrict__ covt,
                                                    const float* __restrict__ covx,
                                                    const float* __restrict__ yfin,
                                                    const float* __restrict__ norms_m,
                                                    float* __restrict__ out) {
  const int c = threadIdx.x;
  const float mt  = mean_t[c];
  const float ctd = covt[c * 257];
  float md = 0.f, cd = 0.f;
  for (int b = 0; b < 8; ++b) {
    const float d = mean_x[b * 256 + c] - mt;
    md += d * d;
    const long long off = (long long)b * 65536LL + c * 257;
    cd += ctd + covx[off] - 2.f * sqrtf(norms_m[b]) * yfin[off];
  }
  __shared__ float sm[256];
  const float md_tot = block_reduce256(md, sm);
  const float cd_tot = block_reduce256(cd, sm);
  if (c == 0) out[0] = (md_tot + cd_tot) * (1.f / 2048.f);
}

// ---------------------------------------------------------------------------
// Orchestration
// ---------------------------------------------------------------------------
extern "C" void kernel_launch(void* const* d_in, const int* in_sizes, int n_in,
                              void* d_out, int out_size, void* d_ws, size_t ws_size,
                              hipStream_t stream) {
  const float* x  = (const float*)d_in[0];   // [8,256,256,256]
  const float* tf = (const float*)d_in[1];   // [256,16384]
  float* out = (float*)d_out;

  float* w = (float*)d_ws;
  float* mean_x  = w;                       // 2048
  float* mean_t  = w + 2048;                // 256
  float* norms_t = w + 2304;                // 1
  float* norms_x = w + 2312;                // 8 (unused downstream)
  float* norms_m = w + 2320;                // 8
  float* covt    = w + 2336;                // 65536
  float* cts     = covt + 65536;            // cov_t_sqrt
  float* yt      = cts  + 65536;
  float* zt      = yt   + 65536;
  float* tt      = zt   + 65536;
  float* yt2     = tt   + 65536;
  float* zt2     = yt2  + 65536;
  float* covx    = zt2  + 65536;            // 8*65536
  float* tmp1    = covx + 524288;
  float* mmat    = tmp1 + 524288;
  float* yb      = mmat + 524288;
  float* zb      = yb   + 524288;
  float* tb      = zb   + 524288;
  float* yb2     = tb   + 524288;
  float* zb2     = yb2  + 524288;

  // 1) row means
  row_mean_kernel<<<2048, 256, 0, stream>>>(x,  mean_x, 65536LL, 1.f / 65536.f);
  row_mean_kernel<<<256,  256, 0, stream>>>(tf, mean_t, 16384LL, 1.f / 16384.f);

  // 2) zero SRM accumulators (finalized in place into covariances)
  zero_kernel<<<2048, 256, 0, stream>>>(covx, 524288LL);
  zero_kernel<<<256,  256, 0, stream>>>(covt, 65536LL);

  // 3) SRMs via split-K bf16 WMMA GEMM with f32 atomics
  gemm_aat_kernel<<<dim3(2, 2, 8 * 16), 256, 0, stream>>>(x,  covx, 65536LL, 16, 4096);
  gemm_aat_kernel<<<dim3(2, 2, 1 * 4),  256, 0, stream>>>(tf, covt, 16384LL, 4,  4096);

  // 4) covariances (+ Frobenius norms)
  finalize_cov_kernel<<<1, 256, 0, stream>>>(covt, mean_t, 1.f / 16384.f, EPS_REG, norms_t);
  finalize_cov_kernel<<<8, 256, 0, stream>>>(covx, mean_x, 1.f / 65536.f, EPS_REG, norms_x);

  // 5) target sqrtm: cov_t_sqrt = sqrtm_ns(cov_t)
  init_yz_kernel<<<dim3(256, 1), 256, 0, stream>>>(covt, norms_t, yt, zt);
  {
    float *cy = yt, *cz = zt, *ny = yt2, *nz = zt2;
    for (int it = 0; it < 12; ++it) {
      gemm256_kernel<<<dim3(2, 2, 1), 256, 0, stream>>>(cz, cy, tt, 0, 0, 0, 1); // t = 1.5I - 0.5 z*y
      gemm256_kernel<<<dim3(2, 2, 1), 256, 0, stream>>>(cy, tt, ny, 0, 0, 0, 0); // y = y*t
      gemm256_kernel<<<dim3(2, 2, 1), 256, 0, stream>>>(tt, cz, nz, 0, 0, 0, 0); // z = t*z
      float* s;
      s = cy; cy = ny; ny = s;
      s = cz; cz = nz; nz = s;
    }
    scale_copy_kernel<<<256, 256, 0, stream>>>(cy, norms_t, cts); // * sqrt(norm)
  }

  // 6) m = cov_t_sqrt @ cov @ cov_t_sqrt   (batched; cts has batch-stride 0)
  gemm256_kernel<<<dim3(2, 2, 8), 256, 0, stream>>>(cts,  covx, tmp1, 0,     65536, 65536, 0);
  gemm256_kernel<<<dim3(2, 2, 8), 256, 0, stream>>>(tmp1, cts,  mmat, 65536, 0,     65536, 0);

  // 7) batched sqrtm_ns(m)
  frob_norm_kernel<<<8, 256, 0, stream>>>(mmat, norms_m);
  init_yz_kernel<<<dim3(256, 8), 256, 0, stream>>>(mmat, norms_m, yb, zb);
  float* yfin = yb;
  {
    float *cy = yb, *cz = zb, *ny = yb2, *nz = zb2;
    for (int it = 0; it < 12; ++it) {
      gemm256_kernel<<<dim3(2, 2, 8), 256, 0, stream>>>(cz, cy, tb, 65536, 65536, 65536, 1);
      gemm256_kernel<<<dim3(2, 2, 8), 256, 0, stream>>>(cy, tb, ny, 65536, 65536, 65536, 0);
      gemm256_kernel<<<dim3(2, 2, 8), 256, 0, stream>>>(tb, cz, nz, 65536, 65536, 65536, 0);
      float* s;
      s = cy; cy = ny; ny = s;
      s = cz; cz = nz; nz = s;
    }
    yfin = cy;
  }

  // 8) scalar loss (folds sqrt(norm_m) rescale into diagonal read)
  final_kernel<<<1, 256, 0, stream>>>(mean_x, mean_t, covt, covx, yfin, norms_m, out);
}
